// xLSTMProj_22668837388797
// MI455X (gfx1250) — matrix operand order repaired and checked by reading, not compiled, not run on hardware
//
#include <hip/hip_runtime.h>

// ---------------------------------------------------------------------------
// Problem constants (must match reference)
// ---------------------------------------------------------------------------
#define EDIM   1024
#define INNER  2048
#define NHQ    512
#define NHEAD  4
#define DH     512
#define KCONV  4
#define SEQ    46
#define BATCH  64
#define OUTD   512
#define NBLK   8
#define TOK    (BATCH * SEQ)        // 2944

typedef __attribute__((ext_vector_type(16))) __bf16 v16bf;
typedef __attribute__((ext_vector_type(8)))  __bf16 v8bf;
typedef __attribute__((ext_vector_type(8)))  float  v8f;
typedef __attribute__((ext_vector_type(4)))  int    v4i;

// ---------------------------------------------------------------------------
// CDNA5 async global->LDS copy (ASYNCcnt-tracked), with portable fallback.
// Probe result: builtin exists and takes generic int4* pointers.
// ---------------------------------------------------------------------------
#if defined(__gfx1250__) && __has_builtin(__builtin_amdgcn_global_load_async_to_lds_b128)
#define HAVE_ASYNC_LDS 1
#else
#define HAVE_ASYNC_LDS 0
#endif

__device__ __forceinline__ void cp16_g2l(void* lds_dst, const void* gsrc) {
#if HAVE_ASYNC_LDS
  __builtin_amdgcn_global_load_async_to_lds_b128((v4i*)(void*)gsrc, (v4i*)lds_dst, 0, 0);
#else
  *(uint4*)lds_dst = *(const uint4*)gsrc;
#endif
}

__device__ __forceinline__ void wait_async_lds() {
#if HAVE_ASYNC_LDS
#if __has_builtin(__builtin_amdgcn_s_wait_asynccnt)
  __builtin_amdgcn_s_wait_asynccnt(0);
#else
  asm volatile("s_wait_asynccnt 0x0" ::: "memory");
#endif
#endif
}

// ---------------------------------------------------------------------------
// bf16 helpers (manual RNE conversion)
// ---------------------------------------------------------------------------
__device__ __forceinline__ __bf16 f32_to_bf16(float f) {
  unsigned u = __float_as_uint(f);
  u = (u + 0x7FFFu + ((u >> 16) & 1u)) >> 16;
  unsigned short s = (unsigned short)u;
  return __builtin_bit_cast(__bf16, s);
}
__device__ __forceinline__ float bf16_to_f32(__bf16 b) {
  unsigned short s = __builtin_bit_cast(unsigned short, b);
  return __uint_as_float(((unsigned)s) << 16);
}
__device__ __forceinline__ float silu_f(float u) {
  return u / (1.0f + __expf(-u));
}

// ---------------------------------------------------------------------------
// Kernel 0: weight transpose + bf16 hi/lo split.
// W[K][N] f32  ->  Wt_hi/Wt_lo [N][K] bf16.  grid = (N/32, K/32), block = 256
// ---------------------------------------------------------------------------
__global__ __launch_bounds__(256) void wsplit_kernel(
    const float* __restrict__ W, __bf16* __restrict__ Wt_hi,
    __bf16* __restrict__ Wt_lo, int Kdim, int Ndim)
{
  __shared__ float tile[32][33];
  const int nb = blockIdx.x * 32, kb = blockIdx.y * 32;
  const int tx = threadIdx.x & 31, ty = threadIdx.x >> 5;   // ty in 0..7
#pragma unroll
  for (int i = 0; i < 4; i++)
    tile[ty + 8 * i][tx] = W[(size_t)(kb + ty + 8 * i) * Ndim + nb + tx];
  __syncthreads();
#pragma unroll
  for (int i = 0; i < 4; i++) {
    float w = tile[tx][ty + 8 * i];
    __bf16 h = f32_to_bf16(w);
    __bf16 l = f32_to_bf16(w - bf16_to_f32(h));
    const size_t o = (size_t)(nb + ty + 8 * i) * Kdim + kb + tx;
    Wt_hi[o] = h;
    Wt_lo[o] = l;
  }
}

// ---------------------------------------------------------------------------
// Kernel 1: per-token LayerNorm, then split result into bf16 hi/lo planes
// grid = TOK, block = 256
// ---------------------------------------------------------------------------
__global__ __launch_bounds__(256) void ln_split_kernel(
    const float* __restrict__ x, const float* __restrict__ ln_w,
    __bf16* __restrict__ a_hi, __bf16* __restrict__ a_lo)
{
  const int t = blockIdx.x, tid = threadIdx.x;
  const float* row = x + (size_t)t * EDIM;
  float v[4], s = 0.f, s2 = 0.f;
#pragma unroll
  for (int i = 0; i < 4; i++) {
    float u = row[tid + 256 * i];
    v[i] = u; s += u; s2 += u * u;
  }
  __shared__ float rs[256], rq[256];
  rs[tid] = s; rq[tid] = s2;
  __syncthreads();
  for (int st = 128; st > 0; st >>= 1) {
    if (tid < st) { rs[tid] += rs[tid + st]; rq[tid] += rq[tid + st]; }
    __syncthreads();
  }
  const float mean = rs[0] * (1.0f / EDIM);
  const float var  = rq[0] * (1.0f / EDIM) - mean * mean;
  const float rstd = rsqrtf(var + 1e-5f);
#pragma unroll
  for (int i = 0; i < 4; i++) {
    int c = tid + 256 * i;
    float y = (v[i] - mean) * rstd * ln_w[c];
    __bf16 hi = f32_to_bf16(y);
    __bf16 lo = f32_to_bf16(y - bf16_to_f32(hi));
    a_hi[(size_t)t * EDIM + c] = hi;
    a_lo[(size_t)t * EDIM + c] = lo;
  }
}

// ---------------------------------------------------------------------------
// Kernel 2: bf16x3 split GEMM  C[M,N] = A[M,K] * W[K,N]  (+ optional residual)
// A: bf16 hi/lo planes, row-major [M][K] -> fragments loaded straight from
//    global (no LDS; rows are wave-private, planes are L2-resident).
// W: pre-split transposed planes Wt[N][K] -> async-copied to double-buffered
//    LDS tiles (pitch 40 -> conflict-free 16B fragment loads).
// Tile: 128(M) x 64(N), K-step 32, 256 threads = 8 wave32.
// grid = (N/64, M/128), block = 256
// ---------------------------------------------------------------------------
#define LP 40
struct AFrag { v16bf h; v16bf l; };

__device__ __forceinline__ AFrag load_afrag(
    const __bf16* __restrict__ Ah, const __bf16* __restrict__ Al,
    size_t rowK, int kbase)
{
  AFrag f;
  v8bf h0 = *(const v8bf*)(Ah + rowK + kbase);
  v8bf h1 = *(const v8bf*)(Ah + rowK + kbase + 16);
  v8bf l0 = *(const v8bf*)(Al + rowK + kbase);
  v8bf l1 = *(const v8bf*)(Al + rowK + kbase + 16);
  f.h = __builtin_shufflevector(h0, h1, 0,1,2,3,4,5,6,7,8,9,10,11,12,13,14,15);
  f.l = __builtin_shufflevector(l0, l1, 0,1,2,3,4,5,6,7,8,9,10,11,12,13,14,15);
  return f;
}

__global__ __launch_bounds__(256) void gemm_bf16x3_kernel(
    const __bf16* __restrict__ Ah, const __bf16* __restrict__ Al,
    const __bf16* __restrict__ Bth, const __bf16* __restrict__ Btl,
    float* __restrict__ Cout, const float* __restrict__ resid,
    int Kdim, int Ndim)
{
  __shared__ __align__(16) __bf16 sBh[2][64 * LP];
  __shared__ __align__(16) __bf16 sBl[2][64 * LP];

  const int tid  = threadIdx.x;
  const int lane = tid & 31;
  const int wv   = tid >> 5;
  const int m0   = blockIdx.y * 128;
  const int n0   = blockIdx.x * 64;

  v8f acc[4];
#pragma unroll
  for (int i = 0; i < 4; i++)
#pragma unroll
    for (int r = 0; r < 8; r++) acc[i][r] = 0.0f;

  // B staging map: thread copies one 16B run per plane
  const int bn  = tid >> 2;            // 0..63 (N row of tile)
  const int bks = (tid & 3) * 8;       // 0,8,16,24 (K offset)
  const size_t bgoff = (size_t)(n0 + bn) * Kdim + bks;
  const int    bloff = bn * LP + bks;

  // fragment addressing
  const int fm  = lane & 15;
  const int akb = (lane >> 4) * 8;     // A frag K base
  const int bkb = (lane >> 4) * 8;     // B frag K base
  const size_t rowK = (size_t)(m0 + 16 * wv + fm) * Kdim;

  // prologue: stage k0 = 0
  cp16_g2l(&sBh[0][bloff], Bth + bgoff);
  cp16_g2l(&sBl[0][bloff], Btl + bgoff);
  AFrag fa = load_afrag(Ah, Al, rowK, akb);
  wait_async_lds();
  __syncthreads();

  int cur = 0;
  for (int k0 = 0; k0 < Kdim; k0 += 32) {
    AFrag fn = fa;
    const bool more = (k0 + 32) < Kdim;
    if (more) {  // stage next tile while computing this one
      cp16_g2l(&sBh[cur ^ 1][bloff], Bth + bgoff + k0 + 32);
      cp16_g2l(&sBl[cur ^ 1][bloff], Btl + bgoff + k0 + 32);
      fn = load_afrag(Ah, Al, rowK, k0 + 32 + akb);
    }

#pragma unroll
    for (int sdx = 0; sdx < 4; sdx++) {
      const int bbase = (sdx * 16 + fm) * LP + bkb;
      v8bf bh0 = *(const v8bf*)&sBh[cur][bbase];
      v8bf bh1 = *(const v8bf*)&sBh[cur][bbase + 16];
      v8bf bl0 = *(const v8bf*)&sBl[cur][bbase];
      v8bf bl1 = *(const v8bf*)&sBl[cur][bbase + 16];
      v16bf bhf = __builtin_shufflevector(bh0, bh1, 0,1,2,3,4,5,6,7,8,9,10,11,12,13,14,15);
      v16bf blf = __builtin_shufflevector(bl0, bl1, 0,1,2,3,4,5,6,7,8,9,10,11,12,13,14,15);
      // bf16x3: Ah*Bh + Ah*Bl + Al*Bh, accumulated in f32
      acc[sdx] = __builtin_amdgcn_wmma_f32_16x16x32_bf16(false, fa.h, false, bhf,
                                                         (short)0, acc[sdx], false, false);
      acc[sdx] = __builtin_amdgcn_wmma_f32_16x16x32_bf16(false, fa.h, false, blf,
                                                         (short)0, acc[sdx], false, false);
      acc[sdx] = __builtin_amdgcn_wmma_f32_16x16x32_bf16(false, fa.l, false, bhf,
                                                         (short)0, acc[sdx], false, false);
    }
    wait_async_lds();
    __syncthreads();
    fa = fn;
    cur ^= 1;
  }

  // epilogue: C layout VGPR r -> M = r + 8*(lane>>4), N = lane&15
  const int mrow0 = m0 + 16 * wv + ((lane >> 4) ? 8 : 0);
  const int ncol0 = n0 + (lane & 15);
#pragma unroll
  for (int sdx = 0; sdx < 4; sdx++) {
#pragma unroll
    for (int r = 0; r < 8; r++) {
      const size_t o = (size_t)(mrow0 + r) * Ndim + ncol0 + 16 * sdx;
      float c = acc[sdx][r];
      if (resid) c += resid[o];
      Cout[o] = c;
    }
  }
}

// ---------------------------------------------------------------------------
// Kernel 3: causal depthwise conv (K=4) + SiLU + head-wise q/k/v (4x4 blocks)
// one thread per (token, head). grid = TOK*NHQ/256, block = 256
// ---------------------------------------------------------------------------
__global__ __launch_bounds__(256) void conv_qkv_kernel(
    const float* __restrict__ up,
    const float* __restrict__ conv_w, const float* __restrict__ conv_b,
    const float* __restrict__ wq, const float* __restrict__ wk,
    const float* __restrict__ wvw,
    float* __restrict__ x_act, float* __restrict__ qb,
    float* __restrict__ kb, float* __restrict__ vb)
{
  const int idx = blockIdx.x * 256 + threadIdx.x;
  const int t = idx >> 9;          // / NHQ
  const int h = idx & 511;
  const int s = t % SEQ;
  const int c0 = 4 * h;

  const float* base = up + (size_t)t * (2 * INNER) + c0;   // x_in slice
  float4 z4 = make_float4(0.f, 0.f, 0.f, 0.f);
  float4 xs0 = *(const float4*)base;                                   // s
  float4 xm1 = (s >= 1) ? *(const float4*)(base - (2 * INNER))     : z4;
  float4 xm2 = (s >= 2) ? *(const float4*)(base - 2 * (2 * INNER)) : z4;
  float4 xm3 = (s >= 3) ? *(const float4*)(base - 3 * (2 * INNER)) : z4;
  const float* p0 = (const float*)&xs0;
  const float* p1 = (const float*)&xm1;
  const float* p2 = (const float*)&xm2;
  const float* p3 = (const float*)&xm3;

  const float* cw = conv_w + (size_t)c0 * KCONV;
  float xa[4];
#pragma unroll
  for (int i = 0; i < 4; i++) {
    float u = cw[i * 4 + 0] * p3[i] + cw[i * 4 + 1] * p2[i] +
              cw[i * 4 + 2] * p1[i] + cw[i * 4 + 3] * p0[i] + conv_b[c0 + i];
    xa[i] = silu_f(u);
  }
  *(float4*)(x_act + (size_t)t * INNER + c0) = make_float4(xa[0], xa[1], xa[2], xa[3]);

  const float* wqh = wq  + (size_t)h * 16;
  const float* wkh = wk  + (size_t)h * 16;
  const float* wvh = wvw + (size_t)h * 16;
  float qo[4], ko[4], vo[4];
#pragma unroll
  for (int o = 0; o < 4; o++) {
    qo[o] = wqh[o*4+0]*xa[0] + wqh[o*4+1]*xa[1] + wqh[o*4+2]*xa[2] + wqh[o*4+3]*xa[3];
    ko[o] = wkh[o*4+0]*xa[0] + wkh[o*4+1]*xa[1] + wkh[o*4+2]*xa[2] + wkh[o*4+3]*xa[3];
    vo[o] = wvh[o*4+0]*p0[0] + wvh[o*4+1]*p0[1] + wvh[o*4+2]*p0[2] + wvh[o*4+3]*p0[3];
  }
  *(float4*)(qb + (size_t)t * INNER + c0) = make_float4(qo[0], qo[1], qo[2], qo[3]);
  *(float4*)(kb + (size_t)t * INNER + c0) = make_float4(ko[0], ko[1], ko[2], ko[3]);
  *(float4*)(vb + (size_t)t * INNER + c0) = make_float4(vo[0], vo[1], vo[2], vo[3]);
}

// ---------------------------------------------------------------------------
// Kernel 4: gate pre-activations  ipre/fpre = concat(q,k,v) @ W + b
// grid = TOK, block = 256 (block reduction over 6144 channels)
// ---------------------------------------------------------------------------
__global__ __launch_bounds__(256) void gates_kernel(
    const float* __restrict__ qb, const float* __restrict__ kb,
    const float* __restrict__ vb,
    const float* __restrict__ ig_w, const float* __restrict__ ig_b,
    const float* __restrict__ fg_w, const float* __restrict__ fg_b,
    float* __restrict__ ipre, float* __restrict__ fpre)
{
  const int t = blockIdx.x, tid = threadIdx.x;
  float accI[4] = {0.f, 0.f, 0.f, 0.f};
  float accF[4] = {0.f, 0.f, 0.f, 0.f};
  for (int c = tid; c < 3 * INNER; c += 256) {
    float val;
    if (c < INNER)          val = qb[(size_t)t * INNER + c];
    else if (c < 2 * INNER) val = kb[(size_t)t * INNER + (c - INNER)];
    else                    val = vb[(size_t)t * INNER + (c - 2 * INNER)];
    float4 wi = *(const float4*)(ig_w + (size_t)c * 4);
    float4 wf = *(const float4*)(fg_w + (size_t)c * 4);
    accI[0] += val * wi.x; accI[1] += val * wi.y;
    accI[2] += val * wi.z; accI[3] += val * wi.w;
    accF[0] += val * wf.x; accF[1] += val * wf.y;
    accF[2] += val * wf.z; accF[3] += val * wf.w;
  }
  __shared__ float red[8][256];
#pragma unroll
  for (int g = 0; g < 4; g++) { red[g][tid] = accI[g]; red[4 + g][tid] = accF[g]; }
  __syncthreads();
  for (int st = 128; st > 0; st >>= 1) {
    if (tid < st)
#pragma unroll
      for (int g = 0; g < 8; g++) red[g][tid] += red[g][tid + st];
    __syncthreads();
  }
  const int b = t / SEQ, s = t % SEQ;
  if (tid < 4)
    ipre[((size_t)b * NHEAD + tid) * SEQ + s] = red[tid][0] + ig_b[tid];
  else if (tid < 8)
    fpre[((size_t)b * NHEAD + (tid - 4)) * SEQ + s] = red[tid][0] + fg_b[tid - 4];
}

// ---------------------------------------------------------------------------
// Kernel 5: mLSTM attention per (batch, head). grid = BATCH*NHEAD, block = 256
// ---------------------------------------------------------------------------
__global__ __launch_bounds__(256) void mlstm_kernel(
    const float* __restrict__ qb, const float* __restrict__ kb,
    const float* __restrict__ vb,
    const float* __restrict__ ipre, const float* __restrict__ fpre,
    float* __restrict__ h_att)
{
  const int bh = blockIdx.x;
  const int b = bh >> 2, h = bh & 3;
  const int tid = threadIdx.x;

  __shared__ float sQ[SEQ * 128];
  __shared__ float sK[SEQ * 128];
  __shared__ float sW[SEQ * 48];
  __shared__ float cum[SEQ], ipr[SEQ];

  if (tid == 0) {
    float c = 0.f;
    for (int s = 0; s < SEQ; s++) {
      float fx = fpre[(size_t)bh * SEQ + s];
      float ls = (fx >= 0.f) ? -log1pf(__expf(-fx)) : fx - log1pf(__expf(fx));
      c += ls;
      cum[s] = c;
      ipr[s] = ipre[(size_t)bh * SEQ + s];
    }
  }
  __syncthreads();

  // causal (t,s) pair assignment
  const int P = SEQ * (SEQ + 1) / 2;     // 1081
  int myT[5], myS[5]; float accD[5]; int np = 0;
  for (int p = tid; p < P; p += 256) {
    int tr = (int)((sqrtf(8.f * (float)p + 1.f) - 1.f) * 0.5f);
    while ((tr + 1) * (tr + 2) / 2 <= p) tr++;
    while (tr * (tr + 1) / 2 > p) tr--;
    myT[np] = tr; myS[np] = p - tr * (tr + 1) / 2; accD[np] = 0.f; np++;
  }

  const size_t gbase = (size_t)b * SEQ * INNER + (size_t)h * DH;
  for (int c0 = 0; c0 < DH; c0 += 128) {
    for (int i = tid; i < SEQ * 128; i += 256) {
      int r = i >> 7, d = i & 127;
      sQ[i] = qb[gbase + (size_t)r * INNER + c0 + d];
      sK[i] = kb[gbase + (size_t)r * INNER + c0 + d];
    }
    __syncthreads();
    for (int j = 0; j < np; j++) {
      const float* qr = &sQ[myT[j] * 128];
      const float* kr = &sK[myS[j] * 128];
      float a = 0.f;
      for (int d = 0; d < 128; d++) a = fmaf(qr[d], kr[d], a);
      accD[j] += a;
    }
    __syncthreads();
  }

  const float scale = 0.044194173824159216f;   // DH^-0.5
  for (int i = tid; i < SEQ * 48; i += 256) sW[i] = 0.f;
  __syncthreads();
  for (int j = 0; j < np; j++) sW[myT[j] * 48 + myS[j]] = accD[j] * scale;
  __syncthreads();

  if (tid < SEQ) {
    const int tr = tid;
    float m = -INFINITY;
    for (int sc = 0; sc <= tr; sc++)
      m = fmaxf(m, cum[tr] - cum[sc] + ipr[sc]);
    float ssum = 0.f;
    for (int sc = 0; sc <= tr; sc++) {
      float ld = cum[tr] - cum[sc] + ipr[sc];
      float Cv = sW[tr * 48 + sc] * __expf(ld - m);
      sW[tr * 48 + sc] = Cv;
      ssum += Cv;
    }
    float inv = 1.f / (fmaxf(fabsf(ssum), __expf(-m)) + 1e-6f);
    for (int sc = 0; sc <= tr; sc++) sW[tr * 48 + sc] *= inv;
  }
  __syncthreads();

  // h[t,d] = sum_{s<=t} W[t,s] * v[s,d]   (reuse sQ as V chunk)
  float* sV = sQ;
  for (int c0 = 0; c0 < DH; c0 += 128) {
    for (int i = tid; i < SEQ * 128; i += 256) {
      int r = i >> 7, d = i & 127;
      sV[i] = vb[gbase + (size_t)r * INNER + c0 + d];
    }
    __syncthreads();
    for (int o = tid; o < SEQ * 128; o += 256) {
      int tr = o >> 7, d = o & 127;
      float a = 0.f;
      for (int sc = 0; sc <= tr; sc++) a = fmaf(sW[tr * 48 + sc], sV[sc * 128 + d], a);
      h_att[gbase + (size_t)tr * INNER + c0 + d] = a;
    }
    __syncthreads();
  }
}

// ---------------------------------------------------------------------------
// Kernel 6: per-(token,head) group-norm + skip*x_act + silu(z) gate,
// then bf16 hi/lo split for the down GEMM. grid = TOK*NHEAD, block = 256
// ---------------------------------------------------------------------------
__global__ __launch_bounds__(256) void postmix_kernel(
    const float* __restrict__ h_att, const float* __restrict__ up,
    const float* __restrict__ x_act,
    const float* __restrict__ mhn_w, const float* __restrict__ skip,
    __bf16* __restrict__ d_hi, __bf16* __restrict__ d_lo)
{
  const int blk = blockIdx.x;
  const int t = blk >> 2, h = blk & 3;
  const int tid = threadIdx.x;
  const float* hr = h_att + (size_t)t * INNER + (size_t)h * DH;
  float v0 = hr[tid], v1 = hr[tid + 256];

  __shared__ float rs[256], rq[256];
  rs[tid] = v0 + v1;
  rq[tid] = v0 * v0 + v1 * v1;
  __syncthreads();
  for (int st = 128; st > 0; st >>= 1) {
    if (tid < st) { rs[tid] += rs[tid + st]; rq[tid] += rq[tid + st]; }
    __syncthreads();
  }
  const float mean = rs[0] * (1.0f / DH);
  const float var  = rq[0] * (1.0f / DH) - mean * mean;
  const float rstd = rsqrtf(var + 1e-5f);

#pragma unroll
  for (int i = 0; i < 2; i++) {
    int d = tid + 256 * i;
    int c = h * DH + d;
    float hv = (i == 0) ? v0 : v1;
    float g = (hv - mean) * rstd * mhn_w[c] + skip[c] * x_act[(size_t)t * INNER + c];
    float z = up[(size_t)t * (2 * INNER) + INNER + c];
    float outv = g * silu_f(z);
    __bf16 hi = f32_to_bf16(outv);
    d_hi[(size_t)t * INNER + c] = hi;
    d_lo[(size_t)t * INNER + c] = f32_to_bf16(outv - bf16_to_f32(hi));
  }
}

// ---------------------------------------------------------------------------
// Kernel 7: final post-LN on last token + projection. grid = BATCH, block = 256
// ---------------------------------------------------------------------------
__global__ __launch_bounds__(256) void final_proj_kernel(
    const float* __restrict__ x_cur, const float* __restrict__ post_ln_w,
    const float* __restrict__ proj_w, const float* __restrict__ proj_b,
    float* __restrict__ out)
{
  const int b = blockIdx.x, tid = threadIdx.x;
  const float* row = x_cur + ((size_t)b * SEQ + (SEQ - 1)) * EDIM;
  float v[4], s = 0.f, s2 = 0.f;
#pragma unroll
  for (int i = 0; i < 4; i++) {
    float u = row[tid + 256 * i];
    v[i] = u; s += u; s2 += u * u;
  }
  __shared__ float rs[256], rq[256];
  __shared__ float sx[EDIM];
  rs[tid] = s; rq[tid] = s2;
  __syncthreads();
  for (int st = 128; st > 0; st >>= 1) {
    if (tid < st) { rs[tid] += rs[tid + st]; rq[tid] += rq[tid + st]; }
    __syncthreads();
  }
  const float mean = rs[0] * (1.0f / EDIM);
  const float var  = rq[0] * (1.0f / EDIM) - mean * mean;
  const float rstd = rsqrtf(var + 1e-5f);
#pragma unroll
  for (int i = 0; i < 4; i++) {
    int c = tid + 256 * i;
    sx[c] = (v[i] - mean) * rstd * post_ln_w[c];
  }
  __syncthreads();
  for (int o = tid; o < OUTD; o += 256) {
    float a = proj_b[o];
    for (int k = 0; k < EDIM; k++) a = fmaf(sx[k], proj_w[(size_t)k * OUTD + o], a);
    out[(size_t)b * OUTD + o] = a;
  }
}

// ---------------------------------------------------------------------------
// Host side
// ---------------------------------------------------------------------------
extern "C" void kernel_launch(void* const* d_in, const int* in_sizes, int n_in,
                              void* d_out, int out_size, void* d_ws, size_t ws_size,
                              hipStream_t stream) {
  (void)in_sizes; (void)n_in; (void)out_size; (void)ws_size;

  const float* x        = (const float*)d_in[0];
  const float* ln_w     = (const float*)d_in[1];
  const float* w_up     = (const float*)d_in[2];
  const float* conv_w   = (const float*)d_in[3];
  const float* conv_b   = (const float*)d_in[4];
  const float* wq       = (const float*)d_in[5];
  const float* wk       = (const float*)d_in[6];
  const float* wvp      = (const float*)d_in[7];
  const float* ig_w     = (const float*)d_in[8];
  const float* ig_b     = (const float*)d_in[9];
  const float* fg_w     = (const float*)d_in[10];
  const float* fg_b     = (const float*)d_in[11];
  const float* skip     = (const float*)d_in[12];
  const float* mhn_w    = (const float*)d_in[13];
  const float* w_down   = (const float*)d_in[14];
  const float* post_ln  = (const float*)d_in[15];
  const float* proj_w   = (const float*)d_in[16];
  const float* proj_b   = (const float*)d_in[17];
  float* out = (float*)d_out;

  char* ws = (char*)d_ws;
  auto take = [&](size_t bytes) -> char* {
    char* p = ws;
    ws += (bytes + 255) & ~(size_t)255;
    return p;
  };
  float*  x_cur  = (float*)take((size_t)TOK * EDIM * 4);
  __bf16* a_hi   = (__bf16*)take((size_t)TOK * EDIM * 2);
  __bf16* a_lo   = (__bf16*)take((size_t)TOK * EDIM * 2);
  float*  up     = (float*)take((size_t)TOK * (2 * INNER) * 4);
  float*  x_act  = (float*)take((size_t)TOK * INNER * 4);
  float*  qb     = (float*)take((size_t)TOK * INNER * 4);
  float*  kbuf   = (float*)take((size_t)TOK * INNER * 4);
  float*  vbuf   = (float*)take((size_t)TOK * INNER * 4);
  float*  ipre   = (float*)take((size_t)BATCH * NHEAD * SEQ * 4);
  float*  fpre   = (float*)take((size_t)BATCH * NHEAD * SEQ * 4);
  float*  h_att  = (float*)take((size_t)TOK * INNER * 4);
  __bf16* d_hi   = (__bf16*)take((size_t)TOK * INNER * 2);
  __bf16* d_lo   = (__bf16*)take((size_t)TOK * INNER * 2);
  __bf16* wtu_hi = (__bf16*)take((size_t)(2 * INNER) * EDIM * 2);
  __bf16* wtu_lo = (__bf16*)take((size_t)(2 * INNER) * EDIM * 2);
  __bf16* wtd_hi = (__bf16*)take((size_t)EDIM * INNER * 2);
  __bf16* wtd_lo = (__bf16*)take((size_t)EDIM * INNER * 2);

  (void)hipMemcpyAsync(x_cur, x, (size_t)TOK * EDIM * 4, hipMemcpyDeviceToDevice, stream);

  for (int blk = 0; blk < NBLK; blk++) {
    // transpose + bf16-split this block's weights (L2-resident afterwards)
    wsplit_kernel<<<dim3((2 * INNER) / 32, EDIM / 32), 256, 0, stream>>>(
        w_up + (size_t)blk * EDIM * (2 * INNER), wtu_hi, wtu_lo, EDIM, 2 * INNER);
    wsplit_kernel<<<dim3(EDIM / 32, INNER / 32), 256, 0, stream>>>(
        w_down + (size_t)blk * INNER * EDIM, wtd_hi, wtd_lo, INNER, EDIM);

    ln_split_kernel<<<TOK, 256, 0, stream>>>(x_cur, ln_w + (size_t)blk * EDIM, a_hi, a_lo);

    gemm_bf16x3_kernel<<<dim3((2 * INNER) / 64, TOK / 128), 256, 0, stream>>>(
        a_hi, a_lo, wtu_hi, wtu_lo, up, nullptr, EDIM, 2 * INNER);

    conv_qkv_kernel<<<(TOK * NHQ) / 256, 256, 0, stream>>>(
        up,
        conv_w + (size_t)blk * INNER * KCONV, conv_b + (size_t)blk * INNER,
        wq + (size_t)blk * NHQ * 16, wk + (size_t)blk * NHQ * 16,
        wvp + (size_t)blk * NHQ * 16,
        x_act, qb, kbuf, vbuf);

    gates_kernel<<<TOK, 256, 0, stream>>>(
        qb, kbuf, vbuf,
        ig_w + (size_t)blk * (3 * INNER) * NHEAD, ig_b + (size_t)blk * NHEAD,
        fg_w + (size_t)blk * (3 * INNER) * NHEAD, fg_b + (size_t)blk * NHEAD,
        ipre, fpre);

    mlstm_kernel<<<BATCH * NHEAD, 256, 0, stream>>>(qb, kbuf, vbuf, ipre, fpre, h_att);

    postmix_kernel<<<TOK * NHEAD, 256, 0, stream>>>(
        h_att, up, x_act,
        mhn_w + (size_t)blk * INNER, skip + (size_t)blk * INNER,
        d_hi, d_lo);

    gemm_bf16x3_kernel<<<dim3(EDIM / 64, TOK / 128), 256, 0, stream>>>(
        d_hi, d_lo, wtd_hi, wtd_lo, x_cur, x_cur, INNER, EDIM);
  }

  final_proj_kernel<<<BATCH, 256, 0, stream>>>(x_cur, post_ln, proj_w, proj_b, out);
}
